// AttentionModule_46823733461763
// MI455X (gfx1250) — matrix operand order repaired and checked by reading, not compiled, not run on hardware
//
#include <hip/hip_runtime.h>

// ---------------------------------------------------------------------------
// AttentionModule for MI455X (gfx1250, wave32, WMMA).
// B=4, H=W=64, C_IN=128, F=256, N=H*W=4096, total rows M=16384.
// All heavy math runs through v_wmma_f32_16x16x32_bf16.
// ---------------------------------------------------------------------------

typedef __attribute__((ext_vector_type(16))) __bf16 v16bf;
typedef __attribute__((ext_vector_type(8)))  __bf16 v8bf;
typedef __attribute__((ext_vector_type(8)))  float  v8f;

#define MROWS  16384   // B*H*W
#define CIN    128
#define FDIM   256
#define NSEQ   4096    // H*W per batch

__device__ __forceinline__ __bf16 to_bf16(float f) {
  unsigned u = __builtin_bit_cast(unsigned, f);
  unsigned short b = (unsigned short)((u + 0x7FFFu + ((u >> 16) & 1u)) >> 16); // RNE
  return __builtin_bit_cast(__bf16, b);
}

__device__ __forceinline__ v8f wmma_bf16(v16bf a, v16bf b, v8f c) {
  // D = A(16x32 bf16) * B(32x16 bf16) + C(16x16 f32)
  return __builtin_amdgcn_wmma_f32_16x16x32_bf16(false, a, false, b, (short)0, c,
                                                 false, false);
}

// A fragment (16x32, MxK): lane L -> row m0 + L%16 ; K(e) = k0 + e + 16*(L/16)
__device__ __forceinline__ v16bf load_a_frag(const __bf16* __restrict__ base, int ld,
                                             int m0, int k0, int laneM, int h) {
  const __bf16* p = base + (long)(m0 + laneM) * ld + k0 + 16 * h;
  v16bf a;
  ((v8bf*)&a)[0] = *(const v8bf*)(p);        // e = 0..7
  ((v8bf*)&a)[1] = *(const v8bf*)(p + 8);    // e = 8..15
  return a;
}

// B fragment (32x16, KxN) from N-major ("B transposed") storage Bt[N][K]:
// lane L -> col n0 + L%16 ; K(e) = k0 + e + 16*(L/16)  -> one contiguous 32B chunk
__device__ __forceinline__ v16bf load_b_frag(const __bf16* __restrict__ base, int ld,
                                             int n0, int k0, int laneM, int h) {
  return *(const v16bf*)(base + (long)(n0 + laneM) * ld + k0 + 16 * h);
}

// ---------------------------------------------------------------------------
// Prep kernels: fp32 -> bf16 conversions / transposes
// ---------------------------------------------------------------------------
__global__ void convert_bf16_kernel(const float* __restrict__ src,
                                    __bf16* __restrict__ dst, long n) {
  long i = (long)blockIdx.x * blockDim.x + threadIdx.x;
  if (i < n) dst[i] = to_bf16(src[i]);
}

// dst[c][r] = src[r][c]  (dst is [C][R])
__global__ void transpose_bf16_kernel(const float* __restrict__ src,
                                      __bf16* __restrict__ dst, int R, int C) {
  int i = blockIdx.x * blockDim.x + threadIdx.x;
  if (i >= R * C) return;
  int r = i / C, c = i % C;
  dst[(long)c * R + r] = to_bf16(src[i]);
}

// ---------------------------------------------------------------------------
// GEMM: C[M,N] = A[M,K] (bf16, row-major) * Bt[N,K] (bf16, N-major) + bias[N]
// One wave computes a 16x64 tile; block = 8 waves = 128 rows x 64 cols.
// Optional outputs: f32, bf16, bf16-transposed-per-batch (for V^T).
// ---------------------------------------------------------------------------
__global__ __launch_bounds__(256)
void gemm_bias_16x64(const __bf16* __restrict__ A, const __bf16* __restrict__ Bt,
                     const float* __restrict__ bias, int K, int N,
                     float* __restrict__ outF, __bf16* __restrict__ outB,
                     __bf16* __restrict__ outT, int batchN) {
  const int lane  = threadIdx.x & 31;
  const int wave  = threadIdx.x >> 5;
  const int laneM = lane & 15;
  const int h     = lane >> 4;
  const int m0    = (blockIdx.x * 8 + wave) * 16;
  const int n0    = blockIdx.y * 64;

  v8f zero = {0.f, 0.f, 0.f, 0.f, 0.f, 0.f, 0.f, 0.f};
  v8f acc[4] = {zero, zero, zero, zero};

  for (int k0 = 0; k0 < K; k0 += 32) {
    v16bf a = load_a_frag(A, K, m0, k0, laneM, h);
#pragma unroll
    for (int j = 0; j < 4; ++j) {
      v16bf b = load_b_frag(Bt, K, n0 + 16 * j, k0, laneM, h);
      acc[j]  = wmma_bf16(a, b, acc[j]);
    }
  }

#pragma unroll
  for (int j = 0; j < 4; ++j) {
    const int col = n0 + 16 * j + laneM;
    const float bv = bias[col];
#pragma unroll
    for (int r = 0; r < 8; ++r) {
      const int row = m0 + r + 8 * h;
      float val = acc[j][r] + bv;
      long idx = (long)row * N + col;
      if (outF) outF[idx] = val;
      if (outB) outB[idx] = to_bf16(val);
      if (outT) {  // outT[batch][col][n]  (V transposed per batch)
        int bidx = row / batchN, nn = row % batchN;
        outT[((long)bidx * N + col) * batchN + nn] = to_bf16(val);
      }
    }
  }
}

// ---------------------------------------------------------------------------
// Fused flash attention + residual:
//   out = gamma * softmax(Q K^T) V + x
// One wave owns 16 query rows with the full 16x256 f32 accumulator in VGPRs;
// streams keys in 64-wide blocks (BN=64 halves softmax/rescale VALU per key
// vs BN=32) with online softmax. P is re-laid-out from C-fragment form to
// A-fragment form through per-wave LDS (16x64 bf16 per wave).
// ---------------------------------------------------------------------------
__global__ __launch_bounds__(256)
void attn_fused_kernel(const __bf16* __restrict__ Q, const __bf16* __restrict__ Kmat,
                       const __bf16* __restrict__ Vt, const float* __restrict__ xres,
                       const float* __restrict__ gamma, float* __restrict__ out) {
  __shared__ __attribute__((aligned(32))) __bf16 pbuf[8][16][64];

  const int lane  = threadIdx.x & 31;
  const int wave  = threadIdx.x >> 5;
  const int laneM = lane & 15;
  const int h     = lane >> 4;
  const int m0    = blockIdx.x * 128 + wave * 16;   // global query row base
  const int batch = m0 >> 12;                       // / NSEQ
  const __bf16* Kb = Kmat + (long)batch * NSEQ * FDIM;  // K rows   [NSEQ][FDIM]
  const __bf16* Vb = Vt   + (long)batch * FDIM * NSEQ;  // V^T rows [FDIM][NSEQ]

  // Preload all Q A-fragments for these 16 rows (16x256 -> 8 frags, 64 VGPRs).
  v16bf qf[8];
#pragma unroll
  for (int i = 0; i < 8; ++i) qf[i] = load_a_frag(Q, FDIM, m0, 32 * i, laneM, h);

  v8f zero = {0.f, 0.f, 0.f, 0.f, 0.f, 0.f, 0.f, 0.f};
  v8f acc[16];
#pragma unroll
  for (int jj = 0; jj < 16; ++jj) acc[jj] = zero;

  float mrow[8], lrow[8];
#pragma unroll
  for (int r = 0; r < 8; ++r) { mrow[r] = -1e30f; lrow[r] = 0.f; }

  for (int kb = 0; kb < NSEQ; kb += 64) {
    if (kb + 64 < NSEQ)  // pull next key tile toward L2/WGP$
      __builtin_prefetch(Kb + (long)(kb + 64) * FDIM, 0, 0);

    // ---- S[16,64] = Q (16x256) * K_blk^T ----
    v8f s[4] = {zero, zero, zero, zero};
#pragma unroll
    for (int i = 0; i < 8; ++i) {
#pragma unroll
      for (int j = 0; j < 4; ++j) {
        v16bf b = load_b_frag(Kb, FDIM, kb + 16 * j, 32 * i, laneM, h);
        s[j] = wmma_bf16(qf[i], b, s[j]);
      }
    }

    // ---- online softmax over this 64-key block ----
    float scale[8];
#pragma unroll
    for (int r = 0; r < 8; ++r) {
      float mx = fmaxf(fmaxf(s[0][r], s[1][r]), fmaxf(s[2][r], s[3][r]));
#pragma unroll
      for (int off = 1; off < 16; off <<= 1) mx = fmaxf(mx, __shfl_xor(mx, off, 32));
      float nm = fmaxf(mrow[r], mx);
      scale[r] = __expf(mrow[r] - nm);
      mrow[r]  = nm;
      float rs = 0.f;
#pragma unroll
      for (int j = 0; j < 4; ++j) {
        float p = __expf(s[j][r] - nm);
        pbuf[wave][r + 8 * h][16 * j + laneM] = to_bf16(p);
        rs += p;
      }
#pragma unroll
      for (int off = 1; off < 16; off <<= 1) rs += __shfl_xor(rs, off, 32);
      lrow[r] = lrow[r] * scale[r] + rs;
    }

#pragma unroll
    for (int jj = 0; jj < 16; ++jj)
#pragma unroll
      for (int r = 0; r < 8; ++r) acc[jj][r] *= scale[r];

    // ---- reload P in A-fragment layout from LDS (same wave; dscnt ordered) ----
    v16bf pa[2];
#pragma unroll
    for (int i2 = 0; i2 < 2; ++i2) {
      const __bf16* pp = &pbuf[wave][laneM][32 * i2 + 16 * h];
      ((v8bf*)&pa[i2])[0] = *(const v8bf*)(pp);
      ((v8bf*)&pa[i2])[1] = *(const v8bf*)(pp + 8);
    }

    // ---- acc[16x256] += P (16x64) * V_blk (64x256) ----
#pragma unroll
    for (int jj = 0; jj < 16; ++jj) {
      v16bf b0 = load_b_frag(Vb, NSEQ, 16 * jj, kb,      laneM, h);
      acc[jj]  = wmma_bf16(pa[0], b0, acc[jj]);
      v16bf b1 = load_b_frag(Vb, NSEQ, 16 * jj, kb + 32, laneM, h);
      acc[jj]  = wmma_bf16(pa[1], b1, acc[jj]);
    }
  }

  // ---- epilogue: out = gamma * (acc / l) + x ----
  const float g = gamma[0];
#pragma unroll
  for (int jj = 0; jj < 16; ++jj)
#pragma unroll
    for (int r = 0; r < 8; ++r) {
      int row = m0 + r + 8 * h;
      int col = 16 * jj + laneM;
      long idx = (long)row * FDIM + col;
      out[idx] = g * (acc[jj][r] / lrow[r]) + xres[idx];
    }
}

// ---------------------------------------------------------------------------
// Host-side orchestration
// ---------------------------------------------------------------------------
extern "C" void kernel_launch(void* const* d_in, const int* in_sizes, int n_in,
                              void* d_out, int out_size, void* d_ws, size_t ws_size,
                              hipStream_t stream) {
  (void)in_sizes; (void)n_in; (void)out_size; (void)ws_size;
  const float* inputs = (const float*)d_in[0];  // [16384,128]
  const float* W_proj = (const float*)d_in[1];  // [128,256]
  const float* b_proj = (const float*)d_in[2];  // [256]
  const float* W_q    = (const float*)d_in[3];  // [256,256]
  const float* b_q    = (const float*)d_in[4];
  const float* W_k    = (const float*)d_in[5];
  const float* b_k    = (const float*)d_in[6];
  const float* W_v    = (const float*)d_in[7];
  const float* b_v    = (const float*)d_in[8];
  const float* gamma  = (const float*)d_in[9];  // [1]
  float* out = (float*)d_out;                   // [16384,256] f32

  // Workspace layout (256B aligned slices), ~38 MB total.
  char* ws = (char*)d_ws;
  size_t off = 0;
  auto take = [&](size_t bytes) -> char* {
    char* p = ws + off;
    off = (off + bytes + 255) & ~(size_t)255;
    return p;
  };
  __bf16* inb  = (__bf16*)take((size_t)MROWS * CIN * 2);   // inputs bf16 [M,128]
  __bf16* wpT  = (__bf16*)take((size_t)FDIM * CIN * 2);    // W_proj^T   [256,128]
  __bf16* wqT  = (__bf16*)take((size_t)FDIM * FDIM * 2);   // W_q^T      [256,256]
  __bf16* wkT  = (__bf16*)take((size_t)FDIM * FDIM * 2);
  __bf16* wvT  = (__bf16*)take((size_t)FDIM * FDIM * 2);
  __bf16* xb   = (__bf16*)take((size_t)MROWS * FDIM * 2);  // x bf16     [M,256]
  __bf16* qb   = (__bf16*)take((size_t)MROWS * FDIM * 2);  // Q bf16     [M,256]
  __bf16* kbuf = (__bf16*)take((size_t)MROWS * FDIM * 2);  // K bf16     [M,256]
  __bf16* vT   = (__bf16*)take((size_t)MROWS * FDIM * 2);  // V^T bf16   [B,256,4096]

  // 0) precision prep
  convert_bf16_kernel<<<(MROWS * CIN + 255) / 256, 256, 0, stream>>>(
      inputs, inb, (long)MROWS * CIN);
  transpose_bf16_kernel<<<(CIN * FDIM + 255) / 256, 256, 0, stream>>>(W_proj, wpT, CIN, FDIM);
  transpose_bf16_kernel<<<(FDIM * FDIM + 255) / 256, 256, 0, stream>>>(W_q, wqT, FDIM, FDIM);
  transpose_bf16_kernel<<<(FDIM * FDIM + 255) / 256, 256, 0, stream>>>(W_k, wkT, FDIM, FDIM);
  transpose_bf16_kernel<<<(FDIM * FDIM + 255) / 256, 256, 0, stream>>>(W_v, wvT, FDIM, FDIM);

  dim3 ggrid(MROWS / 128, FDIM / 64);
  // 1) x = in @ W_proj + b_proj  (f32 copy of x lives in d_out for the residual)
  gemm_bias_16x64<<<ggrid, 256, 0, stream>>>(inb, wpT, b_proj, CIN, FDIM,
                                             out, xb, nullptr, NSEQ);
  // 2) q / k / v = x @ W_* + b_*   (v stored transposed per batch)
  gemm_bias_16x64<<<ggrid, 256, 0, stream>>>(xb, wqT, b_q, FDIM, FDIM,
                                             nullptr, qb, nullptr, NSEQ);
  gemm_bias_16x64<<<ggrid, 256, 0, stream>>>(xb, wkT, b_k, FDIM, FDIM,
                                             nullptr, kbuf, nullptr, NSEQ);
  gemm_bias_16x64<<<ggrid, 256, 0, stream>>>(xb, wvT, b_v, FDIM, FDIM,
                                             nullptr, nullptr, vT, NSEQ);
  // 3) fused attention + residual (reads x from d_out, overwrites d_out)
  attn_fused_kernel<<<MROWS / 128, 256, 0, stream>>>(qb, kbuf, vT, out, gamma, out);
}
